// MPNNConv_919123001903
// MI455X (gfx1250) — compile-verified, matrix operand order checked by reading, not run on hardware
//
#include <hip/hip_runtime.h>

typedef __attribute__((ext_vector_type(16))) _Float16 v16h;
typedef __attribute__((ext_vector_type(8)))  float    v8f;

#ifndef __has_builtin
#define __has_builtin(x) 0
#endif
#if __has_builtin(__builtin_amdgcn_tensor_load_to_lds)
#define HAVE_TDM 1
#else
#define HAVE_TDM 0
#endif

#define N_NODES   10000
#define N_EDGES   640000
#define NODE_DIM  128
#define EDGE_DIM  32
#define IN_DIM    288      // 2*NODE_DIM + EDGE_DIM
#define OUT_DIM   128
#define NT        8        // OUT_DIM / 16 column tiles
#define KS1       9        // IN_DIM  / 32 K-steps, layer 1
#define KS2       4        // OUT_DIM / 32 K-steps, layer 2
#define WAVES     4        // waves per block (one 16-edge tile per wave)
#define STRIDE_A  296      // 288 + 8 halves pad -> 592 B row stride
#define STRIDE_H  136      // 128 + 8 halves pad

// K index inside a 16x32 f16 A/B fragment for half-slot i of `lane`
// (CDNA5 ISA 7.12.2: lanes 0-15 hold K {0..7,16..23}, lanes 16-31 hold K {8..15,24..31})
__device__ __forceinline__ int kmap(int i, int lane) {
  int q = i >> 1, r = i & 1;
  return ((q < 4) ? (q << 1) : (16 + ((q - 4) << 1))) + ((lane & 16) ? 8 : 0) + r;
}

// ---------------- prep: shuffle W1/W2 (f32 [K][N]) into per-lane B fragments (f16)
__global__ void build_frags(const float* __restrict__ W1, const float* __restrict__ W2,
                            _Float16* __restrict__ f1, _Float16* __restrict__ f2) {
  int gid = blockIdx.x * blockDim.x + threadIdx.x;
  const int n1 = NT * KS1 * 512;   // 36864 halves
  const int n2 = NT * KS2 * 512;   // 16384 halves
  if (gid < n1) {
    int nt = gid / (KS1 * 512); int rem = gid % (KS1 * 512);
    int ks = rem / 512;         int lr  = rem % 512;
    int lane = lr >> 4;         int i   = lr & 15;
    f1[gid] = (_Float16)W1[(ks * 32 + kmap(i, lane)) * OUT_DIM + nt * 16 + (lane & 15)];
  } else if (gid < n1 + n2) {
    int g = gid - n1;
    int nt = g / (KS2 * 512);   int rem = g % (KS2 * 512);
    int ks = rem / 512;         int lr  = rem % 512;
    int lane = lr >> 4;         int i   = lr & 15;
    f2[g] = (_Float16)W2[(ks * 32 + kmap(i, lane)) * OUT_DIM + nt * 16 + (lane & 15)];
  }
}

// ---------------- prep: h (f32) -> h16 (f16) table, and zero the scatter target
__global__ void h_to_half(const float* __restrict__ h, _Float16* __restrict__ h16, int n) {
  int i = blockIdx.x * blockDim.x + threadIdx.x;
  if (i < n) h16[i] = (_Float16)h[i];
}

__global__ void zero_out(float* __restrict__ out, int n) {
  int i = blockIdx.x * blockDim.x + threadIdx.x;
  if (i < n) out[i] = 0.0f;
}

#if HAVE_TDM
typedef unsigned __attribute__((ext_vector_type(4))) v4u;
typedef int      __attribute__((ext_vector_type(8))) v8i;
typedef int      __attribute__((ext_vector_type(4))) v4i;

// TDM gather: 16 rows of 128 f16 selected by 16-bit row indices, written to LDS
// with 84-dword padding after each 64-dword (256 B) row -> 592 B LDS row stride.
__device__ __forceinline__ void tdm_gather16(unsigned lds_addr, const _Float16* table,
                                             int myidx, int lane_base) {
  unsigned long long ga = (unsigned long long)(uintptr_t)table;
  v4u g0; v8i g1; v4i g2, g3;
  g0[0] = 0x80000001u;                                  // gather_mode=1, idx16, count=1
  g0[1] = lds_addr;                                     // LDS byte address of tile
  g0[2] = (unsigned)(ga & 0xFFFFFFFFu);                 // global_addr[31:0]
  g0[3] = (unsigned)((ga >> 32) & 0x01FFFFFFu) | (2u << 30);   // addr[56:32] | type=2
  g1[0] = (1 << 16)      // data_size = 2 bytes
        | (1 << 20)      // pad_enable
        | (5 << 22)      // pad_interval: every 64 dwords (one 256 B row)
        | (83 << 25);    // pad_amount: 84 dwords = 336 B  -> 592 B row stride
  g1[1] = (int)(128u << 16);               // tensor_dim0 = 128 elements
  g1[2] = (int)((unsigned)N_NODES << 16);  // tensor_dim1 = 10000 rows
  g1[3] = (int)(128u << 16);               // tile_dim0 = 128 elements
  g1[4] = 16;                              // tile_dim1 = 16 valid indices
  g1[5] = 128;                             // tensor_dim0_stride = 128 elements
  g1[6] = 0; g1[7] = 0;
#pragma unroll
  for (int d = 0; d < 4; ++d) {
    int lo  = __builtin_amdgcn_readlane(myidx, lane_base + 2 * d);
    int hi  = __builtin_amdgcn_readlane(myidx, lane_base + 2 * d + 1);
    int lo2 = __builtin_amdgcn_readlane(myidx, lane_base + 8 + 2 * d);
    int hi2 = __builtin_amdgcn_readlane(myidx, lane_base + 8 + 2 * d + 1);
    g2[d] = (lo & 0xFFFF) | (hi << 16);                 // row indices 0..7
    g3[d] = (lo2 & 0xFFFF) | (hi2 << 16);               // row indices 8..15
  }
  v8i gx = {0, 0, 0, 0, 0, 0, 0, 0};                    // unused/reserved group
  __builtin_amdgcn_tensor_load_to_lds(g0, g1, g2, g3, gx, 0);
}
#endif

// ---------------- main: one 16-edge tile per wave, both layers via v_wmma_f32_16x16x32_f16
__global__ __launch_bounds__(WAVES * 32)
void mpnn_edge_kernel(const _Float16* __restrict__ h16, const long long* __restrict__ eidx,
                      const float* __restrict__ ef, const float* __restrict__ b1,
                      const float* __restrict__ b2, const _Float16* __restrict__ f1,
                      const _Float16* __restrict__ f2, float* __restrict__ out) {
  __shared__ _Float16 ldsA[WAVES][16][STRIDE_A];  // gathered [16 edges x 288] f16
  __shared__ _Float16 ldsH[WAVES][16][STRIDE_H];  // hidden   [16 edges x 128] f16

  const int wave = threadIdx.x >> 5;
  const int lane = threadIdx.x & 31;
  const long long tile = (long long)blockIdx.x * WAVES + wave;
  const long long base = tile * 16;                // first edge of this tile
  const long long* rows = eidx;                    // edge_index[0]
  const long long* cols = eidx + N_EDGES;          // edge_index[1]

  // ---- stage gathered input tile (h16[src] | h16[dst] | ef[e]) into LDS
#if HAVE_TDM
  {
    // lanes 0-15 carry src row ids, lanes 16-31 carry dst row ids
    int myidx = (int)((lane < 16) ? rows[base + lane] : cols[base + (lane - 16)]);
    unsigned lbase = (unsigned)(uintptr_t)&ldsA[wave][0][0];
    tdm_gather16(lbase, h16, myidx, 0);            // src rows -> cols 0..127
    tdm_gather16(lbase + 256u, h16, myidx, 16);    // dst rows -> cols 128..255
  }
#else
  for (int m = 0; m < 16; ++m) {
    const long long e = base + m;
    const int src = (int)rows[e];
    const int dst = (int)cols[e];
    const unsigned long long* ps =
        (const unsigned long long*)(h16 + (size_t)src * NODE_DIM);
    const unsigned long long* pd =
        (const unsigned long long*)(h16 + (size_t)dst * NODE_DIM);
    *(unsigned long long*)&ldsA[wave][m][4 * lane]             = ps[lane];
    *(unsigned long long*)&ldsA[wave][m][NODE_DIM + 4 * lane]  = pd[lane];
  }
#endif
  // edge features: 32 columns, one f32 per lane per edge
  for (int m = 0; m < 16; ++m) {
    const long long e = base + m;
    ldsA[wave][m][2 * NODE_DIM + lane] = (_Float16)ef[e * EDGE_DIM + lane];
  }
#if HAVE_TDM
  __builtin_amdgcn_s_wait_tensorcnt(0);
#endif
  __syncthreads();

  const int mrow  = lane & 15;               // A row / C column-in-tile for this lane
  const int khi   = (lane & 16) ? 8 : 0;     // K offset for high half-wave
  const int mbase = (lane & 16) ? 8 : 0;     // C rows 8..15 for lanes 16..31

  const v16h* F1 = reinterpret_cast<const v16h*>(f1);
  const v16h* F2 = reinterpret_cast<const v16h*>(f2);

  // ---- layer 1: hidden = relu([hs|hd|ef] @ W1 + b1)
  for (int nt = 0; nt < NT; ++nt) {
    float bv = b1[nt * 16 + mrow];
    v8f acc = {bv, bv, bv, bv, bv, bv, bv, bv};
#pragma unroll
    for (int ks = 0; ks < KS1; ++ks) {
      union { v16h v; unsigned u[8]; } a;
      const _Float16* rp = &ldsA[wave][mrow][ks * 32 + khi];
#pragma unroll
      for (int q = 0; q < 8; ++q) {
        int k = (q < 4) ? (q << 1) : (16 + ((q - 4) << 1));
        a.u[q] = *reinterpret_cast<const unsigned*>(rp + k);
      }
      v16h b = F1[(nt * KS1 + ks) * 32 + lane];
      acc = __builtin_amdgcn_wmma_f32_16x16x32_f16(false, a.v, false, b,
                                                   (short)0, acc, false, false);
    }
#pragma unroll
    for (int j = 0; j < 8; ++j) {
      float v = acc[j];
      ldsH[wave][mbase + j][nt * 16 + mrow] = (_Float16)(v > 0.0f ? v : 0.0f);
    }
  }
  __syncthreads();

  // destination rows for this lane's 8 C rows
  long long rowm[8];
#pragma unroll
  for (int j = 0; j < 8; ++j) rowm[j] = rows[base + mbase + j];

  // ---- layer 2: messages = hidden @ W2 + b2, scatter-add to out[row]
  for (int nt = 0; nt < NT; ++nt) {
    float bv = b2[nt * 16 + mrow];
    v8f acc = {bv, bv, bv, bv, bv, bv, bv, bv};
#pragma unroll
    for (int ks = 0; ks < KS2; ++ks) {
      union { v16h v; unsigned u[8]; } a;
      const _Float16* rp = &ldsH[wave][mrow][ks * 32 + khi];
#pragma unroll
      for (int q = 0; q < 8; ++q) {
        int k = (q < 4) ? (q << 1) : (16 + ((q - 4) << 1));
        a.u[q] = *reinterpret_cast<const unsigned*>(rp + k);
      }
      v16h b = F2[(nt * KS2 + ks) * 32 + lane];
      acc = __builtin_amdgcn_wmma_f32_16x16x32_f16(false, a.v, false, b,
                                                   (short)0, acc, false, false);
    }
#pragma unroll
    for (int j = 0; j < 8; ++j) {
      unsafeAtomicAdd(&out[rowm[j] * OUT_DIM + nt * 16 + mrow], acc[j]);
    }
  }
}

extern "C" void kernel_launch(void* const* d_in, const int* in_sizes, int n_in,
                              void* d_out, int out_size, void* d_ws, size_t ws_size,
                              hipStream_t stream) {
  const float*     h  = (const float*)d_in[0];
  const long long* ei = (const long long*)d_in[1];
  const float*     ef = (const float*)d_in[2];
  // d_in[3] is n (scalar, == N_NODES) — baked in as a constant
  const float*     W1 = (const float*)d_in[4];
  const float*     b1 = (const float*)d_in[5];
  const float*     W2 = (const float*)d_in[6];
  const float*     b2 = (const float*)d_in[7];
  float* out = (float*)d_out;

  // workspace layout (all offsets 32 B aligned)
  _Float16* f1  = (_Float16*)d_ws;                         // 36864 halves
  _Float16* f2  = f1 + (size_t)NT * KS1 * 512;             // 16384 halves @ 73728 B
  _Float16* h16 = f2 + (size_t)NT * KS2 * 512;             // 1.28M halves @ 106496 B

  const int frag_total = NT * KS1 * 512 + NT * KS2 * 512;  // 53248
  hipLaunchKernelGGL(build_frags, dim3((frag_total + 255) / 256), dim3(256), 0, stream,
                     W1, W2, f1, f2);
  const int hn = N_NODES * NODE_DIM;                       // 1,280,000
  hipLaunchKernelGGL(h_to_half, dim3((hn + 511) / 512), dim3(512), 0, stream, h, h16, hn);
  hipLaunchKernelGGL(zero_out, dim3((N_NODES * OUT_DIM + 511) / 512), dim3(512), 0, stream,
                     out, N_NODES * OUT_DIM);

  const int tiles = N_EDGES / 16;                          // 40000, exact
  hipLaunchKernelGGL(mpnn_edge_kernel, dim3(tiles / WAVES), dim3(WAVES * 32), 0, stream,
                     h16, ei, ef, b1, b2, f1, f2, out);
}
